// GatedDeltaNet_6124623364301
// MI455X (gfx1250) — matrix operand order repaired and checked by reading, not compiled, not run on hardware
//
#include <hip/hip_runtime.h>
#include <hip/hip_bf16.h>
#include <stdint.h>

// ---------------- model dims ----------------
#define H_SIZE 2048
#define NK 16
#define NV 32
#define DK 128
#define DV 128
#define KEY_DIM 2048        // NK*DK
#define VAL_DIM 4096        // NV*DV
#define QKVZ_DIM 12288      // 2*KEY_DIM + 2*VAL_DIM
#define CONV_DIM 8192       // 2*KEY_DIM + VAL_DIM
#define BATCH 2
#define SEQ 2048
#define ROWS (BATCH * SEQ)  // 4096
#define EPS 1e-6f

// ---------------- WMMA / TDM types ----------------
typedef __attribute__((ext_vector_type(16))) __bf16 v16bf;
typedef __attribute__((ext_vector_type(8)))  float  v8f;
typedef unsigned int v4u __attribute__((ext_vector_type(4)));
typedef int          v8i __attribute__((ext_vector_type(8)));
typedef int          v4i __attribute__((ext_vector_type(4)));

union FragAB { v16bf v; uint4 q[2]; };
union FragC  { v8f   v; float f[8]; };

__device__ __host__ __forceinline__ unsigned short f2bf(float f) {
    union { float f; unsigned int u; } x; x.f = f;
    unsigned int r = x.u + 0x7FFFu + ((x.u >> 16) & 1u);   // RNE
    return (unsigned short)(r >> 16);
}

#if defined(__gfx1250__) && __has_builtin(__builtin_amdgcn_tensor_load_to_lds)
#define USE_TDM 1
#else
#define USE_TDM 0
#endif

// =====================================================================
// f32 -> bf16 bulk conversion (vectorized, n must be multiple of 4)
// =====================================================================
__global__ __launch_bounds__(256)
void cvt_bf16_kernel(const float* __restrict__ in, unsigned short* __restrict__ out,
                     size_t n4)
{
    size_t i = (size_t)blockIdx.x * 256 + threadIdx.x;
    if (i >= n4) return;
    float4 v = ((const float4*)in)[i];
    ushort4 o;
    o.x = f2bf(v.x); o.y = f2bf(v.y); o.z = f2bf(v.z); o.w = f2bf(v.w);
    ((ushort4*)out)[i] = o;
}

// =====================================================================
// bf16-WMMA GEMM: C[M,N] = A[M,K] * B[N,K]^T   (A,B bf16 row-major, C f32)
// Block tile 128x128, K-tile 64 (two 16x16x32 slices), 256 threads =
// 8 wave32 waves (2x4), each wave: 4x2 f32 16x16 accumulators.
// Tile staging: Tensor Data Mover (double-buffered, wave0-issued,
// s_wait_tensorcnt) when available; plain bf16 copies otherwise.
// Requires M%128==0, K%64==0. N guarded (TDM OOB zero-fill / copy guard).
// =====================================================================
#define TM 128
#define TN 128
#define TK 64
#define LDT 72   // LDS row pitch in ushort (64 + 8 pad = 36 DWORDs)

#if USE_TDM
__device__ __forceinline__ void tdm_load_tile(unsigned int lds_addr,
                                              unsigned long long gaddr,
                                              unsigned int rows_avail,
                                              int K_elems)
{
    // D# group 0: count=1 | lds_addr | global_addr[56:0] | type=2
    v4u g0;
    g0[0] = 1u;
    g0[1] = lds_addr;
    g0[2] = (unsigned int)(gaddr & 0xFFFFFFFFull);
    g0[3] = (unsigned int)((gaddr >> 32) & 0x1FFFFFFull) | (2u << 30);
    // D# group 1: data_size=1(2B), pad_enable=1, pad_interval=4(32 DW),
    // pad_amount=3(4 DW) -> LDS row pitch 36 DWORDs = 72 ushorts.
    // tensor_dim0=TK (x always in range), tensor_dim1=rows_avail (OOB rows -> 0),
    // tile_dim0=TK, tile_dim1=128, tensor_dim0_stride=K_elems.
    v8i g1;
    g1[0] = (int)((1u << 16) | (1u << 20) | (4u << 22) | (3u << 25));
    g1[1] = (int)(((unsigned)TK & 0xFFFFu) << 16);
    g1[2] = (int)((((unsigned)TK >> 16) & 0xFFFFu) | ((rows_avail & 0xFFFFu) << 16));
    g1[3] = (int)(((rows_avail >> 16) & 0xFFFFu) | ((unsigned)TK << 16));
    g1[4] = 128;
    g1[5] = (int)(unsigned)K_elems;
    g1[6] = 0;
    g1[7] = 0;
    v4i z4 = {0, 0, 0, 0};
#if __clang_major__ >= 23
    v8i z8 = {0, 0, 0, 0, 0, 0, 0, 0};
    __builtin_amdgcn_tensor_load_to_lds(g0, g1, z4, z4, z8, 0);
#else
    __builtin_amdgcn_tensor_load_to_lds(g0, g1, z4, z4, 0);
#endif
}
#endif

__global__ __launch_bounds__(256)
void wmma_gemm_bf16_kernel(const unsigned short* __restrict__ A,
                           const unsigned short* __restrict__ B,
                           float* __restrict__ C,
                           int M, int N, int K)
{
    __shared__ unsigned short lA[2][TM * LDT];
    __shared__ unsigned short lB[2][TN * LDT];

    const int tid  = threadIdx.x;
    const int lane = tid & 31;
    const int l16  = lane & 15;
    const int lh   = lane >> 4;
    const int wid  = tid >> 5;
    const int wm   = wid & 1;          // 64-row block
    const int wn   = wid >> 1;         // 32-col block
    const int m0   = blockIdx.y * TM;
    const int n0   = blockIdx.x * TN;

    FragC acc[4][2];
#pragma unroll
    for (int i = 0; i < 4; ++i)
#pragma unroll
        for (int j = 0; j < 2; ++j)
#pragma unroll
            for (int r = 0; r < 8; ++r) acc[i][j].f[r] = 0.f;

    const int nkt = K / TK;

#if USE_TDM
    const unsigned long long gA = (unsigned long long)(uintptr_t)(A + (size_t)m0 * K);
    const unsigned long long gB = (unsigned long long)(uintptr_t)(B + (size_t)n0 * K);
    const unsigned int ldsA[2] = { (unsigned int)(uintptr_t)&lA[0][0],
                                   (unsigned int)(uintptr_t)&lA[1][0] };
    const unsigned int ldsB[2] = { (unsigned int)(uintptr_t)&lB[0][0],
                                   (unsigned int)(uintptr_t)&lB[1][0] };
    const unsigned int rowsA = (unsigned int)(M - m0);
    const unsigned int rowsB = (unsigned int)((N > n0) ? (N - n0) : 0);
    if (wid == 0) {                              // prologue: tile 0 -> buf 0
        tdm_load_tile(ldsA[0], gA, rowsA, K);
        tdm_load_tile(ldsB[0], gB, rowsB, K);
    }
#endif

    for (int kt = 0; kt < nkt; ++kt) {
        const int buf = kt & 1;
#if USE_TDM
        if (wid == 0) {
            if (kt + 1 < nkt) {                  // prefetch tile kt+1 via TDM
                unsigned long long off = (unsigned long long)(kt + 1) * TK * 2u;
                tdm_load_tile(ldsA[buf ^ 1], gA + off, rowsA, K);
                tdm_load_tile(ldsB[buf ^ 1], gB + off, rowsB, K);
                __builtin_amdgcn_s_wait_tensorcnt(2);  // tile kt complete
            } else {
                __builtin_amdgcn_s_wait_tensorcnt(0);
            }
        }
        __syncthreads();
#else
        __syncthreads();
        // manual bf16 tile copies (4 x uint4 per thread per tile)
#pragma unroll
        for (int i = 0; i < 4; ++i) {
            int idx = tid + i * 256;
            int row = idx >> 3;
            int c8  = (idx & 7) * 8;
            uint4 v = *(const uint4*)(A + (size_t)(m0 + row) * K + kt * TK + c8);
            *(uint4*)&lA[buf][row * LDT + c8] = v;
        }
#pragma unroll
        for (int i = 0; i < 4; ++i) {
            int idx = tid + i * 256;
            int row = idx >> 3;
            int c8  = (idx & 7) * 8;
            uint4 v = make_uint4(0u, 0u, 0u, 0u);
            if (n0 + row < N)
                v = *(const uint4*)(B + (size_t)(n0 + row) * K + kt * TK + c8);
            *(uint4*)&lB[buf][row * LDT + c8] = v;
        }
        __syncthreads();
#endif
        const unsigned short* tA = lA[buf];
        const unsigned short* tB = lB[buf];
#pragma unroll
        for (int s = 0; s < 2; ++s) {            // two K=32 slices
            FragAB af[4];
#pragma unroll
            for (int i = 0; i < 4; ++i) {
                int m = wm * 64 + i * 16 + l16;
                const unsigned short* p = &tA[m * LDT + s * 32 + lh * 8];
                af[i].q[0] = *(const uint4*)p;
                af[i].q[1] = *(const uint4*)(p + 16);
            }
            FragAB bfr[2];
#pragma unroll
            for (int j = 0; j < 2; ++j) {
                int n = wn * 32 + j * 16 + l16;
                const unsigned short* p = &tB[n * LDT + s * 32 + lh * 8];
                bfr[j].q[0] = *(const uint4*)p;
                bfr[j].q[1] = *(const uint4*)(p + 16);
            }
#pragma unroll
            for (int i = 0; i < 4; ++i)
#pragma unroll
                for (int j = 0; j < 2; ++j)
                    acc[i][j].v = __builtin_amdgcn_wmma_f32_16x16x32_bf16(
                        false, af[i].v, false, bfr[j].v,
                        (short)0, acc[i][j].v, false, false);
        }
#if USE_TDM
        __syncthreads();   // all reads of buf done before TDM overwrites it
#endif
    }

    // store C (C/D layout: n = l16, m = 16*i + lh*8 + r)
#pragma unroll
    for (int i = 0; i < 4; ++i)
#pragma unroll
        for (int j = 0; j < 2; ++j) {
            int n = n0 + wn * 32 + j * 16 + l16;
            if (n < N) {
#pragma unroll
                for (int r = 0; r < 8; ++r) {
                    int m = m0 + wm * 64 + i * 16 + lh * 8 + r;
                    C[(size_t)m * N + n] = acc[i][j].f[r];
                }
            }
        }
}

// =====================================================================
// Causal depthwise conv1d (k=4) + SiLU, in-place, thread per (b,channel)
// =====================================================================
__global__ __launch_bounds__(256)
void conv_silu_kernel(float* __restrict__ qkvz, const float* __restrict__ conv_w)
{
    int tid = blockIdx.x * blockDim.x + threadIdx.x;
    int ch  = tid & (CONV_DIM - 1);
    int b   = tid >> 13;
    float w0 = conv_w[ch * 4 + 0], w1 = conv_w[ch * 4 + 1];
    float w2 = conv_w[ch * 4 + 2], w3 = conv_w[ch * 4 + 3];
    float x0 = 0.f, x1 = 0.f, x2 = 0.f;
    size_t base = (size_t)b * SEQ * QKVZ_DIM + ch;
    for (int t = 0; t < SEQ; ++t) {
        float xt = qkvz[base + (size_t)t * QKVZ_DIM];
        float a  = w0 * x0 + w1 * x1 + w2 * x2 + w3 * xt;
        a = a / (1.f + __expf(-a));
        qkvz[base + (size_t)t * QKVZ_DIM] = a;
        x0 = x1; x1 = x2; x2 = xt;
    }
}

// =====================================================================
// beta = sigmoid(b), ge = exp(-exp(A_log)*softplus(a + dt_bias))
// =====================================================================
__global__ __launch_bounds__(256)
void gate_beta_kernel(const float* __restrict__ ba,
                      const float* __restrict__ A_log,
                      const float* __restrict__ dt_bias,
                      float* __restrict__ ge, float* __restrict__ be)
{
    int i = blockIdx.x * 256 + threadIdx.x;
    int h = i & (NV - 1);
    int r = i >> 5;
    float bv = ba[r * (2 * NV) + h];
    float av = ba[r * (2 * NV) + NV + h];
    be[i] = 1.f / (1.f + __expf(-bv));
    float x  = av + dt_bias[h];
    float sp = (x > 20.f) ? x : log1pf(__expf(x));
    ge[i] = __expf(-__expf(A_log[h]) * sp);
}

// =====================================================================
// Delta-rule scan: WG per (b, head); thread v owns state column (128 VGPRs);
// l2-norm via wave32 __shfl_xor + 4-partial LDS combine (2 barriers/step).
// =====================================================================
__global__ __launch_bounds__(128, 1)
void scan_kernel(const float* __restrict__ qkvz,
                 const float* __restrict__ ge,
                 const float* __restrict__ be,
                 float* __restrict__ core)
{
    __shared__ float ks[DK], qs[DK], rk[4], rq[4];
    const int tid = threadIdx.x;
    const int bh  = blockIdx.x;
    const int b   = bh >> 5;
    const int h   = bh & (NV - 1);
    const int kh  = h >> 1;              // rep = NV/NK = 2

    float st[DK];
#pragma unroll
    for (int i = 0; i < DK; ++i) st[i] = 0.f;

    const size_t tb = (size_t)b * SEQ * QKVZ_DIM;
    for (int t = 0; t < SEQ; ++t) {
        const size_t row = tb + (size_t)t * QKVZ_DIM;
        float qr = qkvz[row + kh * DK + tid];
        float kr = qkvz[row + KEY_DIM + kh * DK + tid];
        float vv = qkvz[row + 2 * KEY_DIM + h * DV + tid];

        float pq = qr * qr, pk = kr * kr;
#pragma unroll
        for (int off = 16; off > 0; off >>= 1) {
            pq += __shfl_xor(pq, off, 32);
            pk += __shfl_xor(pk, off, 32);
        }
        if ((tid & 31) == 0) { rq[tid >> 5] = pq; rk[tid >> 5] = pk; }
        __syncthreads();                               // (A)
        float rnq = rsqrtf(rq[0] + rq[1] + rq[2] + rq[3] + EPS);
        float rnk = rsqrtf(rk[0] + rk[1] + rk[2] + rk[3] + EPS);
        ks[tid] = kr * rnk;
        qs[tid] = qr * rnq;
        __syncthreads();                               // (B)

        const int gi = ((b * SEQ + t) << 5) + h;
        const float g   = ge[gi];
        const float bet = be[gi];

        float kv = 0.f;
#pragma unroll
        for (int kk = 0; kk < DK; kk += 4) {
            float4 k4 = *(const float4*)&ks[kk];
            kv = fmaf(k4.x, st[kk + 0], kv);
            kv = fmaf(k4.y, st[kk + 1], kv);
            kv = fmaf(k4.z, st[kk + 2], kv);
            kv = fmaf(k4.w, st[kk + 3], kv);
        }
        kv *= g;
        const float delta = bet * (vv - kv);

        float o = 0.f;
#pragma unroll
        for (int kk = 0; kk < DK; kk += 4) {
            float4 k4 = *(const float4*)&ks[kk];
            float4 q4 = *(const float4*)&qs[kk];
            st[kk + 0] = fmaf(k4.x, delta, g * st[kk + 0]);
            st[kk + 1] = fmaf(k4.y, delta, g * st[kk + 1]);
            st[kk + 2] = fmaf(k4.z, delta, g * st[kk + 2]);
            st[kk + 3] = fmaf(k4.w, delta, g * st[kk + 3]);
            o = fmaf(q4.x, st[kk + 0], o);
            o = fmaf(q4.y, st[kk + 1], o);
            o = fmaf(q4.z, st[kk + 2], o);
            o = fmaf(q4.w, st[kk + 3], o);
        }
        core[(size_t)(b * SEQ + t) * VAL_DIM + h * DV + tid] = o;
    }
}

// =====================================================================
// RMSNorm * (1 + norm_w) * SiLU(z); writes bf16 so the final WMMA GEMM
// streams a bf16 A matrix directly.
// =====================================================================
__global__ __launch_bounds__(256)
void gate_norm_kernel(const float* __restrict__ core,
                      const float* __restrict__ qkvz,
                      const float* __restrict__ norm_w,
                      unsigned short* __restrict__ core_bf)
{
    __shared__ float red[256];
    const int r   = blockIdx.x;
    const int tid = threadIdx.x;
    const size_t cb = (size_t)r * VAL_DIM;
    const size_t zb = (size_t)r * QKVZ_DIM + (2 * KEY_DIM + VAL_DIM);
    float ss = 0.f;
    for (int i = tid; i < VAL_DIM; i += 256) {
        float v = core[cb + i];
        ss = fmaf(v, v, ss);
    }
    red[tid] = ss; __syncthreads();
    for (int off = 128; off > 0; off >>= 1) {
        if (tid < off) red[tid] += red[tid + off];
        __syncthreads();
    }
    const float rms = rsqrtf(red[0] * (1.f / VAL_DIM) + EPS);
    for (int i = tid; i < VAL_DIM; i += 256) {
        float z  = qkvz[zb + i];
        float sz = z / (1.f + __expf(-z));
        core_bf[cb + i] = f2bf(core[cb + i] * rms * (1.f + norm_w[i]) * sz);
    }
}

// =====================================================================
extern "C" void kernel_launch(void* const* d_in, const int* in_sizes, int n_in,
                              void* d_out, int out_size, void* d_ws, size_t ws_size,
                              hipStream_t stream)
{
    const float* x       = (const float*)d_in[0];
    const float* W_qkvz  = (const float*)d_in[1];
    const float* W_ba    = (const float*)d_in[2];
    const float* conv_w  = (const float*)d_in[3];
    const float* dt_bias = (const float*)d_in[4];
    const float* A_log   = (const float*)d_in[5];
    const float* norm_w  = (const float*)d_in[6];
    const float* W_out   = (const float*)d_in[7];
    float* out = (float*)d_out;

    char* ws = (char*)d_ws;
    size_t off = 0;
    auto carve = [&](size_t bytes) -> void* {
        void* p = ws + off;
        off += (bytes + 255) & ~(size_t)255;
        return p;
    };
    // bf16 operand copies
    unsigned short* xbf   = (unsigned short*)carve((size_t)ROWS * H_SIZE * 2);
    unsigned short* Wqbf  = (unsigned short*)carve((size_t)QKVZ_DIM * H_SIZE * 2);
    unsigned short* Wbabf = (unsigned short*)carve((size_t)2 * NV * H_SIZE * 2);
    unsigned short* Wobf  = (unsigned short*)carve((size_t)H_SIZE * VAL_DIM * 2);
    unsigned short* cbf   = (unsigned short*)carve((size_t)ROWS * VAL_DIM * 2);
    // f32 activations
    float* qkvz = (float*)carve((size_t)ROWS * QKVZ_DIM * 4);
    float* ba   = (float*)carve((size_t)ROWS * 2 * NV * 4);
    float* ge   = (float*)carve((size_t)ROWS * NV * 4);
    float* be   = (float*)carve((size_t)ROWS * NV * 4);
    float* core = (float*)carve((size_t)ROWS * VAL_DIM * 4);

    auto cvt = [&](const float* src, unsigned short* dst, size_t n) {
        size_t n4 = n / 4;
        cvt_bf16_kernel<<<(unsigned)((n4 + 255) / 256), 256, 0, stream>>>(src, dst, n4);
    };
    // 0) bf16 conversions of GEMM operands
    cvt(x,      xbf,   (size_t)ROWS * H_SIZE);
    cvt(W_qkvz, Wqbf,  (size_t)QKVZ_DIM * H_SIZE);
    cvt(W_ba,   Wbabf, (size_t)2 * NV * H_SIZE);
    cvt(W_out,  Wobf,  (size_t)H_SIZE * VAL_DIM);

    // 1) qkvz = x @ W_qkvz^T   (M=4096, N=12288, K=2048)
    {
        dim3 grid(QKVZ_DIM / TN, ROWS / TM);
        wmma_gemm_bf16_kernel<<<grid, 256, 0, stream>>>(xbf, Wqbf, qkvz,
                                                        ROWS, QKVZ_DIM, H_SIZE);
    }
    // 2) ba = x @ W_ba^T       (M=4096, N=64, K=2048)
    {
        dim3 grid(1, ROWS / TM);
        wmma_gemm_bf16_kernel<<<grid, 256, 0, stream>>>(xbf, Wbabf, ba,
                                                        ROWS, 2 * NV, H_SIZE);
    }
    // 3) causal depthwise conv + SiLU (in-place on q/k/v channels)
    conv_silu_kernel<<<(BATCH * CONV_DIM) / 256, 256, 0, stream>>>(qkvz, conv_w);
    // 4) decay ge and beta
    gate_beta_kernel<<<(ROWS * NV) / 256, 256, 0, stream>>>(ba, A_log, dt_bias, ge, be);
    // 5) sequential delta-rule scan, one WG per (b, head)
    scan_kernel<<<BATCH * NV, 128, 0, stream>>>(qkvz, ge, be, core);
    // 6) RMSNorm + gating -> bf16
    gate_norm_kernel<<<ROWS, 256, 0, stream>>>(core, qkvz, norm_w, cbf);
    // 7) out = core @ W_out^T  (M=4096, N=2048, K=4096)
    {
        dim3 grid(H_SIZE / TN, ROWS / TM);
        wmma_gemm_bf16_kernel<<<grid, 256, 0, stream>>>(cbf, Wobf, out,
                                                        ROWS, H_SIZE, VAL_DIM);
    }
    (void)in_sizes; (void)n_in; (void)out_size; (void)ws_size;
}
